// GCNLayer_31980326486285
// MI455X (gfx1250) — matrix operand order repaired and checked by reading, not compiled, not run on hardware
//
#include <hip/hip_runtime.h>
#include <hip/hip_bf16.h>

#define DIM 256
#define KSTEP 32

typedef __attribute__((ext_vector_type(16))) __bf16 v16bf;
typedef __attribute__((ext_vector_type(8)))  float  v8f;

static __device__ __forceinline__ __bf16 f2bf(float f) { return (__bf16)f; }

// ---------------------------------------------------------------- degree
__global__ void gcn_deg_init(float* deg, int n) {
    int i = blockIdx.x * blockDim.x + threadIdx.x;
    if (i < n) deg[i] = 1.0f;              // self-loop contributes 1
}
__global__ void gcn_deg_accum(const int* __restrict__ dst, float* deg, int e) {
    int i = blockIdx.x * blockDim.x + threadIdx.x;
    if (i < e) atomicAdd(&deg[dst[i]], 1.0f);
}
__global__ void gcn_dinv(float* deg, int n) {
    int i = blockIdx.x * blockDim.x + threadIdx.x;
    if (i < n) { float d = deg[i]; deg[i] = (d > 0.0f) ? rsqrtf(d) : 0.0f; }
}

// ---------------------------------------------------------------- converts
__global__ void gcn_cvt_bf16(const float* __restrict__ src, __bf16* __restrict__ dst, int n) {
    int i = blockIdx.x * blockDim.x + threadIdx.x;
    if (i < n) dst[i] = f2bf(src[i]);
}
// W[k][n] (row-major DxD) -> Wt[n][k] bf16 (row-major, k contiguous)
__global__ void gcn_cvt_wt(const float* __restrict__ W, __bf16* __restrict__ Wt) {
    int i = blockIdx.x * blockDim.x + threadIdx.x;   // i = n*DIM + k
    if (i < DIM * DIM) {
        int n = i / DIM, k = i % DIM;
        Wt[i] = f2bf(W[k * DIM + n]);
    }
}

// ---------------------------------------------------------------- WMMA GEMM
// H[N x DIM] = A[N x DIM](bf16,row-major) @ W, with W given as Wt[n][k] bf16.
// Block = 128 threads (4 waves). blockIdx.y = ntile (16 cols), each wave one
// 16-row mtile. Wt slice for the block's 16 columns staged in LDS (8 KB).
__global__ void gcn_gemm_wmma(const __bf16* __restrict__ A,
                              const __bf16* __restrict__ Wt,
                              float* __restrict__ H, int nrows) {
    __shared__ __bf16 sB[16 * DIM];                 // 8192 B
    const int ntile = blockIdx.y;
    const int wave  = threadIdx.x >> 5;
    const int lane  = threadIdx.x & 31;
    const int mtile = blockIdx.x * 4 + wave;

    {   // cooperative LDS fill: 16 rows of Wt (k-contiguous) = 512 uint4
        const uint4* s4 = (const uint4*)(Wt + (size_t)ntile * 16 * DIM);
        uint4* d4 = (uint4*)sB;
        for (int i = threadIdx.x; i < 512; i += 128) d4[i] = s4[i];
    }
    __syncthreads();

    if (mtile * 16 >= nrows) return;

    const int rowA = lane & 15;
    const int kA   = (lane < 16) ? 0 : 8;           // A frag K-base per lane half
    const int colB = lane & 15;
    const int kB   = (lane < 16) ? 0 : 8;           // B frag K-base per lane half
    const __bf16* Arow = A + (size_t)(mtile * 16 + rowA) * DIM;
    const __bf16* Bln  = sB + colB * DIM + kB;

    v8f acc = {};
#pragma unroll
    for (int k0 = 0; k0 < DIM; k0 += KSTEP) {
        v16bf a, b;
        // A 16x32 bf16: lane holds K=[kA..kA+7] and [kA+16..kA+23] of its row
        *(uint4*)&a       = *(const uint4*)(Arow + k0 + kA);
        *((uint4*)&a + 1) = *(const uint4*)(Arow + k0 + kA + 16);
        // B 32x16 bf16: lane holds K=[kB..kB+7] and [kB+16..kB+23] of its column (from Wt)
        *(uint4*)&b       = *(const uint4*)(Bln + k0);
        *((uint4*)&b + 1) = *(const uint4*)(Bln + k0 + 16);
        acc = __builtin_amdgcn_wmma_f32_16x16x32_bf16(
                  false, a, false, b, (short)0, acc, false, false);
    }

    // C/D 16x16 f32: VGPR r -> M = r + (lane<16?0:8), N = lane%16
    float* Hp = H + (size_t)(mtile * 16) * DIM + ntile * 16;
    const int rbase = (lane < 16) ? 0 : 8;
    const int col   = lane & 15;
#pragma unroll
    for (int r = 0; r < 8; ++r)
        Hp[(size_t)(rbase + r) * DIM + col] = acc[r];
}

// ---------------------------------------------------------------- aggregation
// out[i] = bias + dinv[i]^2 * H[i]  (+ resid[i])   -- self-loop + bias (+skip)
__global__ void gcn_agg_init(const float* __restrict__ H, const float* __restrict__ dinv,
                             const float* __restrict__ bias, const float* __restrict__ resid,
                             float* __restrict__ out, int total) {
    int i = blockIdx.x * blockDim.x + threadIdx.x;
    if (i < total) {
        int node = i / DIM, f = i % DIM;
        float di = dinv[node];
        float v = bias[f] + di * di * H[i];
        if (resid) v += resid[i];
        out[i] = v;
    }
}
// one wave per edge: out[dst] += dinv[src]*dinv[dst] * H[src]
__global__ void gcn_agg_edges(const float* __restrict__ H, const float* __restrict__ dinv,
                              const int* __restrict__ src, const int* __restrict__ dst,
                              float* out, int e) {
    int ew   = (blockIdx.x * blockDim.x + threadIdx.x) >> 5;
    int lane = threadIdx.x & 31;
    if (ew >= e) return;
    int s = src[ew], d = dst[ew];
    float nrm = dinv[s] * dinv[d];
    const float* hs = H + (size_t)s * DIM;
    float* od = out + (size_t)d * DIM;
    __builtin_prefetch(hs, 0, 1);                   // global_prefetch_b8
#pragma unroll
    for (int j = 0; j < 8; ++j) {
        int f = lane + 32 * j;                      // coalesced across the wave
        atomicAdd(&od[f], nrm * hs[f]);
    }
}

// ---------------------------------------------------------------- L2 normalize
// one wave per node; writes bf16 (input of second GEMM)
__global__ void gcn_l2norm(const float* __restrict__ in, __bf16* __restrict__ outb, int n) {
    int node = (blockIdx.x * blockDim.x + threadIdx.x) >> 5;
    int lane = threadIdx.x & 31;
    if (node >= n) return;
    const float* p = in + (size_t)node * DIM;
    float v[8], ss = 0.0f;
#pragma unroll
    for (int j = 0; j < 8; ++j) { v[j] = p[lane + 32 * j]; ss += v[j] * v[j]; }
#pragma unroll
    for (int off = 16; off > 0; off >>= 1) ss += __shfl_xor(ss, off, 32);
    float inv = 1.0f / fmaxf(sqrtf(ss), 1e-12f);
    __bf16* q = outb + (size_t)node * DIM;
#pragma unroll
    for (int j = 0; j < 8; ++j) q[lane + 32 * j] = f2bf(v[j] * inv);
}

// ---------------------------------------------------------------- launch
extern "C" void kernel_launch(void* const* d_in, const int* in_sizes, int n_in,
                              void* d_out, int out_size, void* d_ws, size_t ws_size,
                              hipStream_t stream) {
    const float* x  = (const float*)d_in[0];
    const float* W1 = (const float*)d_in[1];
    const float* b1 = (const float*)d_in[2];
    const float* W2 = (const float*)d_in[3];
    const float* b2 = (const float*)d_in[4];
    const int*   ei = (const int*)d_in[5];

    const int N = in_sizes[0] / DIM;
    const int E = in_sizes[5] / 2;
    const int* esrc = ei;
    const int* edst = ei + E;

    // workspace layout (256B aligned)
    char* ws = (char*)d_ws;
    size_t off = 0;
    auto take = [&](size_t bytes) { char* p = ws + off; off = (off + bytes + 255) & ~(size_t)255; return p; };
    float*  dinv = (float*) take((size_t)N * 4);
    __bf16* xb   = (__bf16*)take((size_t)N * DIM * 2);
    __bf16* w1t  = (__bf16*)take((size_t)DIM * DIM * 2);
    __bf16* w2t  = (__bf16*)take((size_t)DIM * DIM * 2);
    float*  Hbuf = (float*) take((size_t)N * DIM * 4);
    float*  agg1 = (float*) take((size_t)N * DIM * 4);
    __bf16* x1b  = (__bf16*)take((size_t)N * DIM * 2);
    (void)ws_size;

    const int total = N * DIM;
    const int B256 = 256;

    // degrees + dinv
    gcn_deg_init <<<(N + B256 - 1) / B256, B256, 0, stream>>>(dinv, N);
    gcn_deg_accum<<<(E + B256 - 1) / B256, B256, 0, stream>>>(edst, dinv, E);
    gcn_dinv     <<<(N + B256 - 1) / B256, B256, 0, stream>>>(dinv, N);

    // bf16 operands
    gcn_cvt_bf16<<<(total + B256 - 1) / B256, B256, 0, stream>>>(x, xb, total);
    gcn_cvt_wt  <<<(DIM * DIM + B256 - 1) / B256, B256, 0, stream>>>(W1, w1t);
    gcn_cvt_wt  <<<(DIM * DIM + B256 - 1) / B256, B256, 0, stream>>>(W2, w2t);

    const int mtiles = N / 16;                       // 10000/16 = 625
    dim3 ggrid((mtiles + 3) / 4, DIM / 16);          // (157, 16), 4 waves/block
    dim3 gblk(128);
    const int eblocks = (E * 32 + B256 - 1) / B256;  // one wave per edge
    const int nblocks = (N * 32 + B256 - 1) / B256;  // one wave per node

    // ---- layer 1
    gcn_gemm_wmma<<<ggrid, gblk, 0, stream>>>(xb, w1t, Hbuf, N);
    gcn_agg_init <<<(total + B256 - 1) / B256, B256, 0, stream>>>(Hbuf, dinv, b1, nullptr, agg1, total);
    gcn_agg_edges<<<eblocks, B256, 0, stream>>>(Hbuf, dinv, esrc, edst, agg1, E);
    gcn_l2norm   <<<nblocks, B256, 0, stream>>>(agg1, x1b, N);

    // ---- layer 2 (+residual) into d_out
    float* out = (float*)d_out;
    gcn_gemm_wmma<<<ggrid, gblk, 0, stream>>>(x1b, w2t, Hbuf, N);
    gcn_agg_init <<<(total + B256 - 1) / B256, B256, 0, stream>>>(Hbuf, dinv, b2, x, out, total);
    gcn_agg_edges<<<eblocks, B256, 0, stream>>>(Hbuf, dinv, esrc, edst, out, E);
    (void)out_size; (void)n_in;
}
